// ScorePredictorRNN_49632642073300
// MI455X (gfx1250) — matrix operand order repaired and checked by reading, not compiled, not run on hardware
//
#include <hip/hip_runtime.h>

typedef __attribute__((ext_vector_type(16))) _Float16 v16h;
typedef __attribute__((ext_vector_type(8)))  float    v8f;

#define B_TOT 2048
#define T_LEN 512
#define IN0   32
#define HID   64
#define G4    256   // 4*HID gate columns
#define BM    16    // batch rows per block
#define HS    72    // h LDS row stride (halves), 144B: 16B aligned, bank-conflict free
#define XS    40    // x LDS row stride (halves), 80B
#define GS    260   // gates LDS row stride (floats): 8*GS % 64 != 0

__device__ __forceinline__ float sigmf(float x) { return 1.0f / (1.0f + __expf(-x)); }
__device__ __forceinline__ float tanh_fast(float x) { return 2.0f / (1.0f + __expf(-2.0f * x)) - 1.0f; }

// A-matrix fragment (16x32 f16), contiguous k order (same as B):
// lanes 0-15: elems 0-15 = K0-15 ; lanes 16-31: K16-31
__device__ __forceinline__ v16h load_A_frag(const _Float16* __restrict__ buf, int stride,
                                            int k0, int lane) {
  const int row = lane & 15;
  const int kb  = (lane >= 16) ? 16 : 0;
  const _Float16* p = buf + row * stride + k0 + kb;
  v16h r;
#pragma unroll
  for (int i = 0; i < 8; ++i) r[i] = p[i];
#pragma unroll
  for (int i = 0; i < 8; ++i) r[8 + i] = p[8 + i];
  return r;
}

// B-matrix fragment (32x16 f16): B(k,n) = W[n][k0+k]; lanes 0-15: K0-15, lanes 16-31: K16-31
__device__ __forceinline__ v16h load_B_frag(const float* __restrict__ W, int ldw,
                                            int n, int k0, int lane) {
  const int kb = (lane >= 16) ? 16 : 0;
  const float* p = W + (size_t)n * ldw + k0 + kb;
  v16h r;
#pragma unroll
  for (int i = 0; i < 16; ++i) r[i] = (_Float16)p[i];
  return r;
}

// C/D 16x16 f32 layout: VGPR r -> lanes 0-15: (M=r, N=lane); lanes 16-31: (M=8+r, N=lane-16)
__device__ __forceinline__ void store_acc(float* __restrict__ g, int nbase, v8f acc, int lane) {
  const int n  = nbase + (lane & 15);
  const int mb = (lane >= 16) ? 8 : 0;
#pragma unroll
  for (int r = 0; r < 8; ++r) g[(mb + r) * GS + n] = acc[r];
}

#define WMMA_F16(a, b, c) \
  __builtin_amdgcn_wmma_f32_16x16x32_f16(false, (a), false, (b), (short)0, (c), false, false)

__global__ __launch_bounds__(256) void lstm2_fused_kernel(
    const float* __restrict__ x,
    const float* __restrict__ W_ih0, const float* __restrict__ W_hh0,
    const float* __restrict__ b_ih0, const float* __restrict__ b_hh0,
    const float* __restrict__ W_ih1, const float* __restrict__ W_hh1,
    const float* __restrict__ b_ih1, const float* __restrict__ b_hh1,
    const float* __restrict__ fc_w,  const float* __restrict__ fc_b,
    float* __restrict__ out)
{
  __shared__ _Float16 h0buf[BM * HS];
  __shared__ _Float16 h1buf[BM * HS];
  __shared__ _Float16 xbuf [BM * XS];
  __shared__ float    gates[BM * GS];
  __shared__ float    bsum0[G4];
  __shared__ float    bsum1[G4];

  const int tid    = threadIdx.x;
  const int lane   = tid & 31;
  const int wave   = tid >> 5;
  const int b_base = blockIdx.x * BM;

  // ---- per-wave weight fragments, loaded once (N-tiles 2*wave, 2*wave+1) ----
  v16h Bih0[2], Bhh0[2][2], Bih1[2][2], Bhh1[2][2];
#pragma unroll
  for (int j = 0; j < 2; ++j) {
    const int n = (wave * 2 + j) * 16 + (lane & 15);
    Bih0[j]    = load_B_frag(W_ih0, IN0, n, 0,  lane);
    Bhh0[j][0] = load_B_frag(W_hh0, HID, n, 0,  lane);
    Bhh0[j][1] = load_B_frag(W_hh0, HID, n, 32, lane);
    Bih1[j][0] = load_B_frag(W_ih1, HID, n, 0,  lane);
    Bih1[j][1] = load_B_frag(W_ih1, HID, n, 32, lane);
    Bhh1[j][0] = load_B_frag(W_hh1, HID, n, 0,  lane);
    Bhh1[j][1] = load_B_frag(W_hh1, HID, n, 32, lane);
  }

  // ---- biases + zero init of recurrent state ----
  bsum0[tid] = b_ih0[tid] + b_hh0[tid];
  bsum1[tid] = b_ih1[tid] + b_hh1[tid];
  for (int i = tid; i < BM * HS; i += 256) {
    h0buf[i] = (_Float16)0.f;
    h1buf[i] = (_Float16)0.f;
  }
  float c0[4] = {0.f, 0.f, 0.f, 0.f};
  float c1[4] = {0.f, 0.f, 0.f, 0.f};
  __syncthreads();

#pragma unroll 1
  for (int t = 0; t < T_LEN; ++t) {
    // stage x tile (BM x IN0) fp32 -> f16 LDS, coalesced 2 floats/thread
    {
      const int idx = tid * 2;               // 512 elements
      const int row = idx >> 5;
      const int col = idx & 31;
      const float* xp = x + ((size_t)(b_base + row) * T_LEN + t) * IN0 + col;
      xbuf[row * XS + col]     = (_Float16)xp[0];
      xbuf[row * XS + col + 1] = (_Float16)xp[1];
    }
    __syncthreads();

    // ---- layer 0 gates: x @ Wih0^T + h0 @ Whh0^T ----
    {
      v16h ax  = load_A_frag(xbuf,  XS, 0,  lane);
      v16h ah0 = load_A_frag(h0buf, HS, 0,  lane);
      v16h ah1 = load_A_frag(h0buf, HS, 32, lane);
#pragma unroll
      for (int j = 0; j < 2; ++j) {
        v8f acc = {};
        acc = WMMA_F16(ax,  Bih0[j],    acc);
        acc = WMMA_F16(ah0, Bhh0[j][0], acc);
        acc = WMMA_F16(ah1, Bhh0[j][1], acc);
        store_acc(gates, (wave * 2 + j) * 16, acc, lane);
      }
    }
    __syncthreads();

    // ---- layer 0 pointwise (PyTorch gate order i,f,g,o) ----
#pragma unroll
    for (int u = 0; u < 4; ++u) {
      const int e   = tid + 256 * u;
      const int row = e >> 6;
      const int j   = e & 63;
      const float gi = gates[row * GS + j]       + bsum0[j];
      const float gf = gates[row * GS + 64 + j]  + bsum0[64 + j];
      const float gg = gates[row * GS + 128 + j] + bsum0[128 + j];
      const float go = gates[row * GS + 192 + j] + bsum0[192 + j];
      const float iv = sigmf(gi), fv = sigmf(gf);
      const float gv = tanh_fast(gg), ov = sigmf(go);
      c0[u] = fv * c0[u] + iv * gv;
      h0buf[row * HS + j] = (_Float16)(ov * tanh_fast(c0[u]));
    }
    __syncthreads();

    // ---- layer 1 gates: h1_t(x) @ Wih1^T + h1_state @ Whh1^T ----
    {
      v16h ax0 = load_A_frag(h0buf, HS, 0,  lane);
      v16h ax1 = load_A_frag(h0buf, HS, 32, lane);
      v16h ah0 = load_A_frag(h1buf, HS, 0,  lane);
      v16h ah1 = load_A_frag(h1buf, HS, 32, lane);
#pragma unroll
      for (int j = 0; j < 2; ++j) {
        v8f acc = {};
        acc = WMMA_F16(ax0, Bih1[j][0], acc);
        acc = WMMA_F16(ax1, Bih1[j][1], acc);
        acc = WMMA_F16(ah0, Bhh1[j][0], acc);
        acc = WMMA_F16(ah1, Bhh1[j][1], acc);
        store_acc(gates, (wave * 2 + j) * 16, acc, lane);
      }
    }
    __syncthreads();

    // ---- layer 1 pointwise ----
#pragma unroll
    for (int u = 0; u < 4; ++u) {
      const int e   = tid + 256 * u;
      const int row = e >> 6;
      const int j   = e & 63;
      const float gi = gates[row * GS + j]       + bsum1[j];
      const float gf = gates[row * GS + 64 + j]  + bsum1[64 + j];
      const float gg = gates[row * GS + 128 + j] + bsum1[128 + j];
      const float go = gates[row * GS + 192 + j] + bsum1[192 + j];
      const float iv = sigmf(gi), fv = sigmf(gf);
      const float gv = tanh_fast(gg), ov = sigmf(go);
      c1[u] = fv * c1[u] + iv * gv;
      h1buf[row * HS + j] = (_Float16)(ov * tanh_fast(c1[u]));
    }
    __syncthreads();
  }

  // ---- final FC: out[b] = h2_last @ fc_w^T + fc_b  (16 rows x 2 outputs) ----
  if (tid < 32) {
    const int row = tid >> 1;
    const int o   = tid & 1;
    float s = fc_b[o];
#pragma unroll
    for (int j = 0; j < HID; ++j)
      s += (float)h1buf[row * HS + j] * fc_w[o * HID + j];
    out[(size_t)(b_base + row) * 2 + o] = s;
  }
}

extern "C" void kernel_launch(void* const* d_in, const int* in_sizes, int n_in,
                              void* d_out, int out_size, void* d_ws, size_t ws_size,
                              hipStream_t stream) {
  const float* x     = (const float*)d_in[0];
  const float* W_ih0 = (const float*)d_in[1];
  const float* W_hh0 = (const float*)d_in[2];
  const float* b_ih0 = (const float*)d_in[3];
  const float* b_hh0 = (const float*)d_in[4];
  const float* W_ih1 = (const float*)d_in[5];
  const float* W_hh1 = (const float*)d_in[6];
  const float* b_ih1 = (const float*)d_in[7];
  const float* b_hh1 = (const float*)d_in[8];
  const float* fc_w  = (const float*)d_in[9];
  const float* fc_b  = (const float*)d_in[10];
  float* out = (float*)d_out;

  dim3 grid(B_TOT / BM);   // 128 blocks
  dim3 block(256);         // 8 wave32s
  hipLaunchKernelGGL(lstm2_fused_kernel, grid, block, 0, stream,
                     x, W_ih0, W_hh0, b_ih0, b_hh0,
                     W_ih1, W_hh1, b_ih1, b_hh1, fc_w, fc_b, out);
}
